// WindowAttention_86904368267307
// MI455X (gfx1250) — matrix operand order repaired and checked by reading, not compiled, not run on hardware
//
#include <hip/hip_runtime.h>
#include <hip/hip_bf16.h>

typedef __attribute__((ext_vector_type(16))) __bf16 v16bf;
typedef __attribute__((ext_vector_type(8)))  float  v8f;
typedef unsigned short u16_t;

#define B_IMG 128
#define SEQ_N 256
#define DIM_C 512
#define HEADS 16
#define HDIM  32
#define ROWS_TOTAL (B_IMG * SEQ_N)      // 32768
#define QSCALE 0.17677669529663689f     // 1/sqrt(32)

// ---------- helpers ----------

__device__ __forceinline__ u16_t f2bf(float x) {
    unsigned u = __float_as_uint(x);
    unsigned r = u + 0x7FFFu + ((u >> 16) & 1u);   // RNE
    return (u16_t)(r >> 16);
}

// A-matrix fragment (16x32 bf16): lane l<16 -> row l, K {0..7,16..23};
// lanes 16..31 -> row l, K {8..15,24..31}. Source is row-major [row, K].
__device__ __forceinline__ v16bf ld_frag_a(const u16_t* __restrict__ p, int stride,
                                           int row_base, int col_base, int lane) {
    int l = lane & 15, h = lane >> 4;
    const u16_t* r = p + (size_t)(row_base + l) * stride + col_base;
    union { uint4 u[2]; v16bf v; } f;
    f.u[0] = *(const uint4*)(r + (h ? 8 : 0));
    f.u[1] = *(const uint4*)(r + (h ? 24 : 16));
    return f.v;
}

// B-matrix fragment (32x16 bf16): lane l holds column l; half0 K {0..7,16..23},
// half1 K {8..15,24..31}. Source row-major: row = B-column, col = K.
__device__ __forceinline__ v16bf ld_frag_b(const u16_t* __restrict__ p, int stride,
                                           int col_base, int k_base, int lane) {
    int l = lane & 15, h = lane >> 4;
    const u16_t* r = p + (size_t)(col_base + l) * stride + k_base;
    union { uint4 u[2]; v16bf v; } f;
    f.u[0] = *(const uint4*)(r + (h ? 8 : 0));
    f.u[1] = *(const uint4*)(r + (h ? 24 : 16));
    return f.v;
}

__device__ __forceinline__ v8f wmma_bf16(v16bf a, v16bf b, v8f c) {
    return __builtin_amdgcn_wmma_f32_16x16x32_bf16(false, a, false, b, (short)0, c, false, false);
}

__device__ __forceinline__ float rmax16(float v) {
    v = fmaxf(v, __shfl_xor(v, 1, 16));
    v = fmaxf(v, __shfl_xor(v, 2, 16));
    v = fmaxf(v, __shfl_xor(v, 4, 16));
    v = fmaxf(v, __shfl_xor(v, 8, 16));
    return v;
}
__device__ __forceinline__ float rsum16(float v) {
    v += __shfl_xor(v, 1, 16);
    v += __shfl_xor(v, 2, 16);
    v += __shfl_xor(v, 4, 16);
    v += __shfl_xor(v, 8, 16);
    return v;
}

// ---------- kernels ----------

// fp32 -> bf16, two elements per thread
__global__ void cvt_bf16_kernel(const float* __restrict__ s, u16_t* __restrict__ d, int n) {
    int i = (blockIdx.x * blockDim.x + threadIdx.x) * 2;
    if (i < n) {
        float2 v = *(const float2*)(s + i);
        d[i]     = f2bf(v.x);
        d[i + 1] = f2bf(v.y);
    }
}

// bias[h][n][m] = rpb_table[rpi(n,m)][h], wh=8 ww=32 (MWH=8, MWW=32)
__global__ void bias_build_kernel(const float* __restrict__ rpb, float* __restrict__ bias) {
    int m = threadIdx.x;
    int n = blockIdx.x;
    int h = blockIdx.y;
    int dh = (n >> 5) - (m >> 5) + 7;       // + MWH-1
    int dw = (n & 31) - (m & 31) + 31;      // + MWW-1
    int idx = dh * 63 + dw;                 // *(2*MWW-1)
    bias[((size_t)h * SEQ_N + n) * SEQ_N + m] = rpb[idx * HEADS + h];
}

// qkv = x @ W^T + b ; per-wave 32x64 tile, 2-deep register pipeline.
// Scatter into q(scaled), k [b,h,n,32] and vT [b,h,32,n].
__global__ __launch_bounds__(256) void qkv_gemm_kernel(
        const u16_t* __restrict__ xb, const u16_t* __restrict__ w,
        const float* __restrict__ bqkv,
        u16_t* __restrict__ qb, u16_t* __restrict__ kbuf, u16_t* __restrict__ vtb) {
    int lane = threadIdx.x & 31, wv = threadIdx.x >> 5;
    int row0 = blockIdx.x * 256 + wv * 32;
    int j0   = blockIdx.y * 64;
    v8f acc[2][4] = {};
    v16bf af[2][2], bf[2][4];
#pragma unroll
    for (int i = 0; i < 2; ++i) af[0][i] = ld_frag_a(xb, DIM_C, row0 + 16 * i, 0, lane);
#pragma unroll
    for (int t = 0; t < 4; ++t) bf[0][t] = ld_frag_b(w, DIM_C, j0 + 16 * t, 0, lane);
#pragma unroll 4
    for (int kk = 0; kk < 16; ++kk) {
        int cur = kk & 1, nxt = cur ^ 1;
        if (kk < 15) {
#pragma unroll
            for (int i = 0; i < 2; ++i)
                af[nxt][i] = ld_frag_a(xb, DIM_C, row0 + 16 * i, (kk + 1) * 32, lane);
#pragma unroll
            for (int t = 0; t < 4; ++t)
                bf[nxt][t] = ld_frag_b(w, DIM_C, j0 + 16 * t, (kk + 1) * 32, lane);
        }
#pragma unroll
        for (int i = 0; i < 2; ++i)
#pragma unroll
            for (int t = 0; t < 4; ++t)
                acc[i][t] = wmma_bf16(af[cur][i], bf[cur][t], acc[i][t]);
    }
    int l = lane & 15, h = lane >> 4;
#pragma unroll
    for (int i = 0; i < 2; ++i) {
#pragma unroll
        for (int t = 0; t < 4; ++t) {
            int j = j0 + t * 16 + l;
            int sec = j >> 9;            // 0=q 1=k 2=v (tile never straddles)
            int jj = j & 511;
            int head = jj >> 5, dd = jj & 31;
            float bv = bqkv[j];
#pragma unroll
            for (int r = 0; r < 8; ++r) {
                int row = row0 + 16 * i + r + 8 * h;
                int bimg = row >> 8, n = row & 255;
                float v = acc[i][t][r] + bv;
                if (sec == 0)
                    qb[(((size_t)(bimg * HEADS + head) * SEQ_N + n) * HDIM) + dd] = f2bf(v * QSCALE);
                else if (sec == 1)
                    kbuf[(((size_t)(bimg * HEADS + head) * SEQ_N + n) * HDIM) + dd] = f2bf(v);
                else
                    vtb[(((size_t)(bimg * HEADS + head) * HDIM + dd) * SEQ_N) + n] = f2bf(v);
            }
        }
    }
}

// flash attention: one wave = 16 query rows of one (b,h); online softmax over
// 8 key blocks of 32; K frags pipelined, V frags hoisted ahead of softmax;
// P re-shaped C/D -> A layout through a per-wave LDS strip.
__global__ __launch_bounds__(256) void attn_kernel(
        const u16_t* __restrict__ qb, const u16_t* __restrict__ kbuf,
        const u16_t* __restrict__ vtb, const float* __restrict__ bias,
        u16_t* __restrict__ ao) {
    __shared__ __align__(16) u16_t Pbuf[8][16][32];
    int lane = threadIdx.x & 31, wv = threadIdx.x >> 5;
    int bh = blockIdx.x >> 1;
    int q0 = (blockIdx.x & 1) * 128 + wv * 16;
    int head = bh & (HEADS - 1), bimg = bh >> 4;
    const u16_t* qp = qb   + (size_t)bh * SEQ_N * HDIM;
    const u16_t* kp = kbuf + (size_t)bh * SEQ_N * HDIM;
    const u16_t* vp = vtb  + (size_t)bh * HDIM * SEQ_N;
    const float* bp = bias + (size_t)head * SEQ_N * SEQ_N;
    int l = lane & 15, h = lane >> 4;

    v16bf qf = ld_frag_a(qp, HDIM, q0, 0, lane);
    v8f oa = {}, ob = {};
    float mrow[8], lrow[8];
#pragma unroll
    for (int r = 0; r < 8; ++r) { mrow[r] = -1e30f; lrow[r] = 0.f; }

    v16bf kf0 = ld_frag_b(kp, HDIM, 0, 0, lane);
    v16bf kf1 = ld_frag_b(kp, HDIM, 16, 0, lane);

    for (int t = 0; t < 8; ++t) {
        int kb0 = t * 32;
        v8f z = {};
        v8f s0 = wmma_bf16(qf, kf0, z);
        v8f s1 = wmma_bf16(qf, kf1, z);
        // hoisted: V frags for this block, K frags for next block
        v16bf va = ld_frag_b(vp, SEQ_N, 0,  kb0, lane);   // dims 0..15
        v16bf vb = ld_frag_b(vp, SEQ_N, 16, kb0, lane);   // dims 16..31
        if (t < 7) {
            kf0 = ld_frag_b(kp, HDIM, kb0 + 32, 0, lane);
            kf1 = ld_frag_b(kp, HDIM, kb0 + 48, 0, lane);
        }
#pragma unroll
        for (int r = 0; r < 8; ++r) {
            int row = q0 + r + 8 * h;
            float x0 = s0[r] + bp[(size_t)row * SEQ_N + kb0 + l];
            float x1 = s1[r] + bp[(size_t)row * SEQ_N + kb0 + 16 + l];
            float mx = rmax16(fmaxf(x0, x1));
            float mn = fmaxf(mrow[r], mx);
            float corr = __expf(mrow[r] - mn);
            float p0 = __expf(x0 - mn);
            float p1 = __expf(x1 - mn);
            float rs = rsum16(p0 + p1);
            lrow[r] = lrow[r] * corr + rs;
            mrow[r] = mn;
            oa[r] *= corr; ob[r] *= corr;
            Pbuf[wv][r + 8 * h][l]      = f2bf(p0);
            Pbuf[wv][r + 8 * h][16 + l] = f2bf(p1);
        }
        __syncthreads();   // uniform; orders LDS strip stores before b128 reload
        v16bf pf = ld_frag_a(&Pbuf[wv][0][0], 32, 0, 0, lane);
        oa = wmma_bf16(pf, va, oa);
        ob = wmma_bf16(pf, vb, ob);
        __syncthreads();   // uniform; protects strip against next-iter overwrite
    }
#pragma unroll
    for (int r = 0; r < 8; ++r) {
        float inv = 1.0f / lrow[r];
        int row = q0 + r + 8 * h;
        size_t base = ((size_t)bimg * SEQ_N + row) * DIM_C + head * HDIM;
        ao[base + l]      = f2bf(oa[r] * inv);
        ao[base + 16 + l] = f2bf(ob[r] * inv);
    }
}

// out = attn_out @ proj_w^T + proj_b (fp32 out); 32x64 per wave, pipelined.
__global__ __launch_bounds__(256) void proj_gemm_kernel(
        const u16_t* __restrict__ aob, const u16_t* __restrict__ w,
        const float* __restrict__ pb, float* __restrict__ out) {
    int lane = threadIdx.x & 31, wv = threadIdx.x >> 5;
    int row0 = blockIdx.x * 256 + wv * 32;
    int j0   = blockIdx.y * 64;
    v8f acc[2][4] = {};
    v16bf af[2][2], bf[2][4];
#pragma unroll
    for (int i = 0; i < 2; ++i) af[0][i] = ld_frag_a(aob, DIM_C, row0 + 16 * i, 0, lane);
#pragma unroll
    for (int t = 0; t < 4; ++t) bf[0][t] = ld_frag_b(w, DIM_C, j0 + 16 * t, 0, lane);
#pragma unroll 4
    for (int kk = 0; kk < 16; ++kk) {
        int cur = kk & 1, nxt = cur ^ 1;
        if (kk < 15) {
#pragma unroll
            for (int i = 0; i < 2; ++i)
                af[nxt][i] = ld_frag_a(aob, DIM_C, row0 + 16 * i, (kk + 1) * 32, lane);
#pragma unroll
            for (int t = 0; t < 4; ++t)
                bf[nxt][t] = ld_frag_b(w, DIM_C, j0 + 16 * t, (kk + 1) * 32, lane);
        }
#pragma unroll
        for (int i = 0; i < 2; ++i)
#pragma unroll
            for (int t = 0; t < 4; ++t)
                acc[i][t] = wmma_bf16(af[cur][i], bf[cur][t], acc[i][t]);
    }
    int l = lane & 15, h = lane >> 4;
#pragma unroll
    for (int i = 0; i < 2; ++i) {
#pragma unroll
        for (int t = 0; t < 4; ++t) {
            int j = j0 + t * 16 + l;
            float bv = pb[j];
#pragma unroll
            for (int r = 0; r < 8; ++r) {
                int row = row0 + 16 * i + r + 8 * h;
                out[(size_t)row * DIM_C + j] = acc[i][t][r] + bv;
            }
        }
    }
}

// ---------- launcher ----------

extern "C" void kernel_launch(void* const* d_in, const int* in_sizes, int n_in,
                              void* d_out, int out_size, void* d_ws, size_t ws_size,
                              hipStream_t stream) {
    const float* x      = (const float*)d_in[0];
    // d_in[1] = q_global (unused by reference)
    const float* qkv_w  = (const float*)d_in[2];
    const float* qkv_b  = (const float*)d_in[3];
    const float* proj_w = (const float*)d_in[4];
    const float* proj_b = (const float*)d_in[5];
    const float* rpb    = (const float*)d_in[6];

    char* ws = (char*)d_ws;
    const size_t NX  = (size_t)ROWS_TOTAL * DIM_C;        // 16,777,216
    const size_t NWQ = (size_t)3 * DIM_C * DIM_C;         // 786,432
    const size_t NWP = (size_t)DIM_C * DIM_C;             // 262,144
    const size_t NQKV = (size_t)B_IMG * HEADS * SEQ_N * HDIM; // 16,777,216

    size_t off = 0;
    u16_t* xb    = (u16_t*)(ws + off); off += NX  * 2;
    u16_t* wqkvb = (u16_t*)(ws + off); off += NWQ * 2;
    u16_t* wprjb = (u16_t*)(ws + off); off += NWP * 2;
    u16_t* qb    = (u16_t*)(ws + off); off += NQKV * 2;
    u16_t* kbuf  = (u16_t*)(ws + off); off += NQKV * 2;
    u16_t* vtb   = (u16_t*)(ws + off); off += NQKV * 2;
    float* biasb = (float*)(ws + off); off += (size_t)HEADS * SEQ_N * SEQ_N * 4;
    u16_t* aob   = (u16_t*)(ws + off); off += NX * 2;

    cvt_bf16_kernel<<<(int)(NX  / 512), 256, 0, stream>>>(x,      xb,    (int)NX);
    cvt_bf16_kernel<<<(int)(NWQ / 512), 256, 0, stream>>>(qkv_w,  wqkvb, (int)NWQ);
    cvt_bf16_kernel<<<(int)(NWP / 512), 256, 0, stream>>>(proj_w, wprjb, (int)NWP);

    bias_build_kernel<<<dim3(SEQ_N, HEADS), SEQ_N, 0, stream>>>(rpb, biasb);

    qkv_gemm_kernel<<<dim3(ROWS_TOTAL / 256, (3 * DIM_C) / 64), 256, 0, stream>>>(
        xb, wqkvb, qkv_b, qb, kbuf, vtb);

    attn_kernel<<<B_IMG * HEADS * 2, 256, 0, stream>>>(qb, kbuf, vtb, biasb, aob);

    proj_gemm_kernel<<<dim3(ROWS_TOTAL / 256, DIM_C / 64), 256, 0, stream>>>(
        aob, wprjb, proj_b, (float*)d_out);
}
